// SortGRU_5617817223317
// MI455X (gfx1250) — matrix operand order repaired and checked by reading, not compiled, not run on hardware
//
#include <hip/hip_runtime.h>
#include <hip/hip_bf16.h>

typedef _Float16 f16;
typedef _Float16 v16h __attribute__((ext_vector_type(16)));
typedef _Float16 v8h  __attribute__((ext_vector_type(8)));
typedef float    v8f  __attribute__((ext_vector_type(8)));

// ---- dims ----
static constexpr int B      = 256;
static constexpr int LSEQ   = 1024;
static constexpr int CINC   = 16;
static constexpr int WINW   = 8;
static constexpr int H      = 256;
static constexpr int NLAY   = 6;
static constexpr int SSTEPS = 128;   // L / WIN
static constexpr int OSEQ   = 4;
static constexpr int OONE   = 4;
static constexpr int DG     = 8;

// ---- ws layout (f16 elements) ----
static constexpr size_t OFF_WIH0 = 0;                       // 768*128
static constexpr size_t OFF_WIH  = OFF_WIH0 + 768 * 128;    // 5*768*256
static constexpr size_t OFF_WHH  = OFF_WIH + 5 * 768 * 256; // 6*768*256
static constexpr size_t OFF_W1   = OFF_WHH + 6 * 768 * 256; // 6*256*256
static constexpr size_t OFF_W2   = OFF_W1 + 6 * 256 * 256;  // 6*128*128
static constexpr size_t OFF_W3   = OFF_W2 + 6 * 128 * 128;  // 6*128*128
static constexpr size_t OFF_W4   = OFF_W3 + 6 * 128 * 128;  // 6*256*512

// ---------------------------------------------------------------- helpers
__device__ __forceinline__ v8f vzero8() {
  v8f z;
#pragma unroll
  for (int i = 0; i < 8; ++i) z[i] = 0.0f;
  return z;
}

// A fragment (16x32 f16) from LDS, row-major [16][stride].
// element j of v16h -> K = (j/8)*16 + hl*8 + (j%8), M = lane&15.
__device__ __forceinline__ v16h frag_a_lds(const f16* base, int stride, int kb, int lane) {
  const int m = lane & 15, hl = lane >> 4;
  const f16* p = base + m * stride + kb + hl * 8;
  v8h lo = *(const v8h*)(p);
  v8h hi = *(const v8h*)(p + 16);
  v16h r;
#pragma unroll
  for (int i = 0; i < 8; ++i) { r[i] = lo[i]; r[i + 8] = hi[i]; }
  return r;
}

// B fragment (32x16 f16) from global weight W[Nout][K] row-major.
// B[k][n] = W[n0+n][k]; element j -> K = kb + (j/8)*16 + hl*8 + (j%8).
__device__ __forceinline__ v16h frag_b_gbl(const f16* __restrict__ W, int K, int n0, int kb, int lane) {
  const int n = lane & 15, hl = lane >> 4;
  const f16* p = W + (size_t)(n0 + n) * K + kb + hl * 8;
  v8h lo = *(const v8h*)(p);
  v8h hi = *(const v8h*)(p + 16);
  v16h r;
#pragma unroll
  for (int i = 0; i < 8; ++i) { r[i] = lo[i]; r[i + 8] = hi[i]; }
  return r;
}

__device__ __forceinline__ v8f wmma16(v16h a, v16h b, v8f c) {
  // (neg_a, A, neg_b, B, c_mod, C, reuse_a, reuse_b)
  return __builtin_amdgcn_wmma_f32_16x16x32_f16(false, a, false, b, (short)0, c, false, false);
}

// C tile (m = r + hl*8, n = lane&15) -> f32 LDS row-major [16][stride]
__device__ __forceinline__ void store_c(float* p, int stride, int n0, v8f c, int lane) {
  const int n = lane & 15, hl = lane >> 4;
#pragma unroll
  for (int r = 0; r < 8; ++r) p[(r + hl * 8) * stride + n0 + n] = c[r];
}

__device__ __forceinline__ float sigf(float x) { return 1.0f / (1.0f + __expf(-x)); }

// ---------------------------------------------------------------- f32 -> f16
__global__ void cvt_f32_f16(const float* __restrict__ src, f16* __restrict__ dst, int n) {
  int i = blockIdx.x * 256 + threadIdx.x;
  if (i < n) dst[i] = (f16)src[i];
}

// ---------------------------------------------------------------- main recurrent kernel
// grid: 16 workgroups (16 batch rows each), block: 256 (8 wave32 waves)
__global__ __launch_bounds__(256, 1)
void sortgru_recurrent(const float* __restrict__ x,
                       const f16* __restrict__ wih0,
                       const f16* __restrict__ wih,
                       const f16* __restrict__ whh,
                       const float* __restrict__ bih,
                       const float* __restrict__ bhh,
                       const float* __restrict__ h0,
                       const f16* __restrict__ w1,
                       const f16* __restrict__ w2,
                       const f16* __restrict__ w3,
                       const f16* __restrict__ w4,
                       float* __restrict__ enc) {
  // 112 KB of CDNA5 WGP LDS (320 KB available)
  __shared__ __align__(16) f16   sHST[NLAY * 16 * 256]; // persistent hidden state, 48 KB
  __shared__ __align__(16) f16   sAIN[16 * 512];        // layer input / shuffled A staging, 16 KB
  __shared__ __align__(16) f16   sH[16 * 256];          // post-GRU h (attn input), 8 KB
  __shared__ __align__(16) f16   sSORT[16 * 256];       // sorted h, 8 KB
  __shared__ __align__(16) float sF32[16 * 512];        // attn f32 staging, 32 KB

  const int tid  = threadIdx.x;
  const int lane = tid & 31;
  const int wave = tid >> 5;
  const int nlo  = lane & 15;
  const int hl   = lane >> 4;
  const int b0   = blockIdx.x * 16;

  // hidden_0 broadcast into resident state
  for (int i = tid; i < NLAY * 16 * 256; i += 256) {
    int l = i >> 12;
    int col = i & 255;
    sHST[i] = (f16)h0[l * 256 + col];
  }
  __syncthreads();

  for (int s = 0; s < SSTEPS; ++s) {
    // x window -> layer-0 input: xs[s,b,j] = x[b, s*8 + j/16, j%16] (contiguous 128 floats)
    for (int i = tid; i < 16 * 128; i += 256) {
      int m = i >> 7, j = i & 127;
      sAIN[m * 512 + j] = (f16)x[(size_t)(b0 + m) * (LSEQ * CINC) + (size_t)s * 128 + j];
    }
    __syncthreads();

    for (int l = 0; l < NLAY; ++l) {
      const int Kin = (l == 0) ? 128 : 256;
      const f16* Wi = (l == 0) ? wih0 : (wih + (size_t)(l - 1) * 768 * 256);
      const f16* Wh = whh + (size_t)l * 768 * 256;
      const float* bi = bih + l * 768;
      const float* bh = bhh + l * 768;
      f16* hst = sHST + l * 16 * 256;
      const int nb = wave * 32; // this wave's 32 h-columns

      // ---------------- GRU: 12 register-resident accumulator tiles ----------------
      v8f gi[3][2], gh[3][2];
#pragma unroll
      for (int g = 0; g < 3; ++g)
#pragma unroll
        for (int t = 0; t < 2; ++t) { gi[g][t] = vzero8(); gh[g][t] = vzero8(); }

      for (int kb = 0; kb < Kin; kb += 32) {
        v16h a = frag_a_lds(sAIN, 512, kb, lane);
#pragma unroll
        for (int g = 0; g < 3; ++g)
#pragma unroll
          for (int t = 0; t < 2; ++t)
            gi[g][t] = wmma16(a, frag_b_gbl(Wi, Kin, g * 256 + nb + t * 16, kb, lane), gi[g][t]);
      }
      for (int kb = 0; kb < 256; kb += 32) {
        v16h a = frag_a_lds(hst, 256, kb, lane);
#pragma unroll
        for (int g = 0; g < 3; ++g)
#pragma unroll
          for (int t = 0; t < 2; ++t)
            gh[g][t] = wmma16(a, frag_b_gbl(Wh, 256, g * 256 + nb + t * 16, kb, lane), gh[g][t]);
      }
      __syncthreads(); // all waves done reading sAIN / hst

      // fused gate nonlinearities, in C-tile layout
#pragma unroll
      for (int t = 0; t < 2; ++t) {
        const int col = nb + t * 16 + nlo;
        const float bir = bi[col], biz = bi[256 + col], bin = bi[512 + col];
        const float bhr = bh[col], bhz = bh[256 + col], bhn = bh[512 + col];
#pragma unroll
        for (int r = 0; r < 8; ++r) {
          const int m = r + hl * 8;
          float rg = sigf((gi[0][t][r] + bir) + (gh[0][t][r] + bhr));
          float zg = sigf((gi[1][t][r] + biz) + (gh[1][t][r] + bhz));
          float ng = tanhf((gi[2][t][r] + bin) + rg * (gh[2][t][r] + bhn));
          float hp = (float)hst[m * 256 + col];
          float hn = (1.0f - zg) * ng + zg * hp;
          sH[m * 256 + col] = (f16)hn;
          sSORT[m * 256 + col] = (f16)hn;
        }
      }
      __syncthreads();

      // ---------------- bitonic sort of each row (ascending, 256 elems) ----------------
      for (int k = 2; k <= 256; k <<= 1) {
        for (int j = k >> 1; j > 0; j >>= 1) {
          const int ixj = tid ^ j;
          if (ixj > tid) {
            const bool up = ((tid & k) == 0);
            for (int row = 0; row < 16; ++row) {
              float a = (float)sSORT[row * 256 + tid];
              float c = (float)sSORT[row * 256 + ixj];
              if ((a > c) == up) {
                sSORT[row * 256 + tid] = (f16)c;
                sSORT[row * 256 + ixj] = (f16)a;
              }
            }
          }
          __syncthreads();
        }
      }

      // ---------------- SortAttn GEMM1: [h; sorted] @ W1.T -> [16][512] ----------------
      {
        v8f acc[4];
#pragma unroll
        for (int t = 0; t < 4; ++t) acc[t] = vzero8();
        const f16* Asrc = (wave < 4) ? sH : sSORT;
        const int c0 = (wave * 64) & 255;
        const f16* Wg = w1 + (size_t)l * 256 * 256;
        for (int kb = 0; kb < 256; kb += 32) {
          v16h a = frag_a_lds(Asrc, 256, kb, lane);
#pragma unroll
          for (int t = 0; t < 4; ++t)
            acc[t] = wmma16(a, frag_b_gbl(Wg, 256, c0 + t * 16, kb, lane), acc[t]);
        }
#pragma unroll
        for (int t = 0; t < 4; ++t) store_c(sF32, 512, wave * 64 + t * 16, acc[t], lane);
      }
      __syncthreads();
      // shuffle ([4,128] -> transpose -> [4,128]) + f16
      for (int i = tid; i < 16 * 512; i += 256) {
        int row = i >> 9, f = i & 511;
        sAIN[i] = (f16)sF32[row * 512 + ((f & 3) * 128 + (f >> 2))];
      }
      __syncthreads();

      // ---------------- GEMM2: per-group @ W2.T ----------------
      {
        v8f acc[4];
#pragma unroll
        for (int t = 0; t < 4; ++t) acc[t] = vzero8();
        const int g = wave >> 1;
        const int nb2 = (wave & 1) * 64;
        const f16* Asrc = sAIN + g * 128;
        const f16* Wg = w2 + (size_t)l * 128 * 128;
        for (int kb = 0; kb < 128; kb += 32) {
          v16h a = frag_a_lds(Asrc, 512, kb, lane);
#pragma unroll
          for (int t = 0; t < 4; ++t)
            acc[t] = wmma16(a, frag_b_gbl(Wg, 128, nb2 + t * 16, kb, lane), acc[t]);
        }
#pragma unroll
        for (int t = 0; t < 4; ++t) store_c(sF32, 512, g * 128 + nb2 + t * 16, acc[t], lane);
      }
      __syncthreads();
      for (int i = tid; i < 16 * 512; i += 256) {
        int row = i >> 9, f = i & 511;
        sAIN[i] = (f16)sF32[row * 512 + ((f & 3) * 128 + (f >> 2))];
      }
      __syncthreads();

      // ---------------- GEMM3: per-group @ W3.T, ReLU ----------------
      {
        v8f acc[4];
#pragma unroll
        for (int t = 0; t < 4; ++t) acc[t] = vzero8();
        const int g = wave >> 1;
        const int nb2 = (wave & 1) * 64;
        const f16* Asrc = sAIN + g * 128;
        const f16* Wg = w3 + (size_t)l * 128 * 128;
        for (int kb = 0; kb < 128; kb += 32) {
          v16h a = frag_a_lds(Asrc, 512, kb, lane);
#pragma unroll
          for (int t = 0; t < 4; ++t)
            acc[t] = wmma16(a, frag_b_gbl(Wg, 128, nb2 + t * 16, kb, lane), acc[t]);
        }
#pragma unroll
        for (int t = 0; t < 4; ++t) {
#pragma unroll
          for (int r = 0; r < 8; ++r) acc[t][r] = fmaxf(acc[t][r], 0.0f);
          store_c(sF32, 512, g * 128 + nb2 + t * 16, acc[t], lane);
        }
      }
      __syncthreads();
      // flatten (identity) -> f16 A staging for W4
      for (int i = tid; i < 16 * 512; i += 256) sAIN[i] = (f16)sF32[i];
      __syncthreads();

      // ---------------- GEMM4: [16][512] @ W4.T -> [16][256]; h * sigmoid ----------------
      {
        v8f acc[2];
#pragma unroll
        for (int t = 0; t < 2; ++t) acc[t] = vzero8();
        const f16* Wg = w4 + (size_t)l * 256 * 512;
        for (int kb = 0; kb < 512; kb += 32) {
          v16h a = frag_a_lds(sAIN, 512, kb, lane);
#pragma unroll
          for (int t = 0; t < 2; ++t)
            acc[t] = wmma16(a, frag_b_gbl(Wg, 512, nb + t * 16, kb, lane), acc[t]);
        }
        __syncthreads(); // all waves done reading sAIN before it is overwritten
#pragma unroll
        for (int t = 0; t < 2; ++t) {
          const int col = nb + t * 16 + nlo;
#pragma unroll
          for (int r = 0; r < 8; ++r) {
            const int m = r + hl * 8;
            float ho = (float)sH[m * 256 + col] * sigf(acc[t][r]);
            hst[m * 256 + col] = (f16)ho;       // next-timestep hidden state
            sAIN[m * 512 + col] = (f16)ho;      // next layer's input
            if (l == NLAY - 1)
              enc[((size_t)(b0 + m) * SSTEPS + s) * H + col] = ho;
          }
        }
      }
      __syncthreads();
    } // layers
  }   // steps
}

// ---------------------------------------------------------------- SeqDecoder: one thread per (b,s)
__global__ void seq_decoder(const float* __restrict__ enc,
                            const float* __restrict__ W1, const float* __restrict__ b1,
                            const float* __restrict__ W2, const float* __restrict__ b2,
                            const float* __restrict__ W3, const float* __restrict__ b3,
                            const float* __restrict__ W4, const float* __restrict__ b4,
                            float* __restrict__ yseq) {
  int idx = blockIdx.x * 256 + threadIdx.x;
  if (idx >= B * SSTEPS) return;
  int b = idx / SSTEPS, s = idx % SSTEPS;
  const float* e = enc + (size_t)idx * H;

  float a1[8][4], a2[8][4], a3[8][4], tbuf[8][4];
  for (int d = 0; d < 8; ++d)
    for (int q = 0; q < 4; ++q) {
      float sum = b1[q];
      for (int j = 0; j < 32; ++j) sum += e[d * 32 + j] * W1[q * 32 + j];
      a1[d][q] = sum;
    }
  for (int p = 0; p < 8; ++p)
    for (int q = 0; q < 4; ++q) { int f = p * 4 + q; tbuf[p][q] = a1[f % 8][f / 8]; }
  for (int d = 0; d < 8; ++d)
    for (int q = 0; q < 4; ++q) {
      float sum = b2[q];
      for (int j = 0; j < 4; ++j) sum += tbuf[d][j] * W2[q * 4 + j];
      a2[d][q] = sum;
    }
  for (int p = 0; p < 8; ++p)
    for (int q = 0; q < 4; ++q) { int f = p * 4 + q; tbuf[p][q] = a2[f % 8][f / 8]; }
  for (int d = 0; d < 8; ++d)
    for (int q = 0; q < 4; ++q) {
      float sum = b3[q];
      for (int j = 0; j < 4; ++j) sum += tbuf[d][j] * W3[q * 4 + j];
      a3[d][q] = sum;
    }
  float c[4][8];
  for (int o = 0; o < 4; ++o)
    for (int g = 0; g < 8; ++g) { int f = o * 8 + g; c[o][g] = a3[f / 4][f % 4]; }
  for (int o = 0; o < 4; ++o)
    for (int w = 0; w < 8; ++w) {
      float sum = b4[w];
      for (int g = 0; g < 8; ++g) sum += c[o][g] * W4[w * 8 + g];
      yseq[((size_t)b * LSEQ + (size_t)s * WINW + w) * OSEQ + o] = sum;
    }
}

// ---------------------------------------------------------------- OneDecoder: one thread per b
__global__ void one_decoder(const float* __restrict__ enc,
                            const float* __restrict__ W1, const float* __restrict__ b1,
                            const float* __restrict__ W2, const float* __restrict__ b2,
                            const float* __restrict__ W3, const float* __restrict__ b3,
                            const float* __restrict__ W4, const float* __restrict__ b4,
                            float* __restrict__ yone) {
  int b = blockIdx.x * 256 + threadIdx.x;
  if (b >= B) return;
  float accum[4] = {0.0f, 0.0f, 0.0f, 0.0f};
  for (int s = SSTEPS - 10; s < SSTEPS; ++s) {
    const float* e = enc + ((size_t)b * SSTEPS + s) * H;
    float a1[8][4], a2[8][4], a3[8][4], tbuf[8][4];
    for (int d = 0; d < 8; ++d)
      for (int q = 0; q < 4; ++q) {
        float sum = b1[q];
        for (int j = 0; j < 32; ++j) sum += e[d * 32 + j] * W1[q * 32 + j];
        a1[d][q] = sum;
      }
    for (int p = 0; p < 8; ++p)
      for (int q = 0; q < 4; ++q) { int f = p * 4 + q; tbuf[p][q] = a1[f % 8][f / 8]; }
    for (int d = 0; d < 8; ++d)
      for (int q = 0; q < 4; ++q) {
        float sum = b2[q];
        for (int j = 0; j < 4; ++j) sum += tbuf[d][j] * W2[q * 4 + j];
        a2[d][q] = sum;
      }
    for (int p = 0; p < 8; ++p)
      for (int q = 0; q < 4; ++q) { int f = p * 4 + q; tbuf[p][q] = a2[f % 8][f / 8]; }
    for (int d = 0; d < 8; ++d)
      for (int q = 0; q < 4; ++q) {
        float sum = b3[q];
        for (int j = 0; j < 4; ++j) sum += tbuf[d][j] * W3[q * 4 + j];
        a3[d][q] = fmaxf(sum, 0.0f);
      }
    for (int o = 0; o < 4; ++o) {
      float sum = b4[o];
      for (int f = 0; f < 32; ++f) sum += a3[f / 4][f % 4] * W4[o * 32 + f];
      accum[o] += 1.0f / (1.0f + __expf(-sum));
    }
  }
  for (int o = 0; o < 4; ++o) yone[b * OONE + o] = accum[o] * 0.1f;
}

// ---------------------------------------------------------------- launch
extern "C" void kernel_launch(void* const* d_in, const int* in_sizes, int n_in,
                              void* d_out, int out_size, void* d_ws, size_t ws_size,
                              hipStream_t stream) {
  const float* x    = (const float*)d_in[0];
  const float* Wih0 = (const float*)d_in[1];
  const float* Wih  = (const float*)d_in[2];
  const float* Whh  = (const float*)d_in[3];
  const float* bih  = (const float*)d_in[4];
  const float* bhh  = (const float*)d_in[5];
  const float* h0   = (const float*)d_in[6];
  const float* aW1  = (const float*)d_in[7];
  const float* aW2  = (const float*)d_in[8];
  const float* aW3  = (const float*)d_in[9];
  const float* aW4  = (const float*)d_in[10];
  const float* sdW1 = (const float*)d_in[11];
  const float* sdb1 = (const float*)d_in[12];
  const float* sdW2 = (const float*)d_in[13];
  const float* sdb2 = (const float*)d_in[14];
  const float* sdW3 = (const float*)d_in[15];
  const float* sdb3 = (const float*)d_in[16];
  const float* sdW4 = (const float*)d_in[17];
  const float* sdb4 = (const float*)d_in[18];
  const float* odW1 = (const float*)d_in[19];
  const float* odb1 = (const float*)d_in[20];
  const float* odW2 = (const float*)d_in[21];
  const float* odb2 = (const float*)d_in[22];
  const float* odW3 = (const float*)d_in[23];
  const float* odb3 = (const float*)d_in[24];
  const float* odW4 = (const float*)d_in[25];
  const float* odb4 = (const float*)d_in[26];

  f16* ws = (f16*)d_ws;
  f16* wih0_h = ws + OFF_WIH0;
  f16* wih_h  = ws + OFF_WIH;
  f16* whh_h  = ws + OFF_WHH;
  f16* w1_h   = ws + OFF_W1;
  f16* w2_h   = ws + OFF_W2;
  f16* w3_h   = ws + OFF_W3;
  f16* w4_h   = ws + OFF_W4;

  auto cvt = [&](const float* src, f16* dst, int n) {
    cvt_f32_f16<<<(n + 255) / 256, 256, 0, stream>>>(src, dst, n);
  };
  cvt(Wih0, wih0_h, 768 * 128);
  cvt(Wih,  wih_h,  5 * 768 * 256);
  cvt(Whh,  whh_h,  6 * 768 * 256);
  cvt(aW1,  w1_h,   6 * 256 * 256);
  cvt(aW2,  w2_h,   6 * 128 * 128);
  cvt(aW3,  w3_h,   6 * 128 * 128);
  cvt(aW4,  w4_h,   6 * 256 * 512);

  float* enc  = (float*)d_out;                    // [B, S, H]
  float* yseq = enc + (size_t)B * SSTEPS * H;     // [B, L, OSEQ]
  float* yone = yseq + (size_t)B * LSEQ * OSEQ;   // [B, OONE]

  sortgru_recurrent<<<B / 16, 256, 0, stream>>>(x, wih0_h, wih_h, whh_h, bih, bhh, h0,
                                                w1_h, w2_h, w3_h, w4_h, enc);
  seq_decoder<<<(B * SSTEPS + 255) / 256, 256, 0, stream>>>(enc, sdW1, sdb1, sdW2, sdb2,
                                                            sdW3, sdb3, sdW4, sdb4, yseq);
  one_decoder<<<1, 256, 0, stream>>>(enc, odW1, odb1, odW2, odb2, odW3, odb3, odW4, odb4, yone);
}
